// Mamba_22497038696540
// MI455X (gfx1250) — matrix operand order repaired and checked by reading, not compiled, not run on hardware
//
#include <hip/hip_runtime.h>
#include <hip/hip_bf16.h>
#include <math.h>

#define DMODEL   512
#define NLAYERS  4
#define DINNER   1024
#define DCONV    8
#define DSTATE   16
#define DTRANK   32
#define SB       4
#define SL       1024
#define NTOK     (SB * SL)          // 4096 tokens

typedef _Float16     v16h __attribute__((ext_vector_type(16)));
typedef _Float16     v4h  __attribute__((ext_vector_type(4)));
typedef float        v8f  __attribute__((ext_vector_type(8)));
typedef unsigned int v4u  __attribute__((ext_vector_type(4)));
typedef int          v4i  __attribute__((ext_vector_type(4)));

union F16Frag { v16h v; v4u u[2]; };

__device__ __forceinline__ float sigmoid_fast(float x)  { return 1.0f / (1.0f + __expf(-x)); }
__device__ __forceinline__ float softplus_fast(float x) { return (x > 20.0f) ? x : log1pf(__expf(x)); }

// ---------------------------------------------------------------------------
// gfx1250 async global->LDS path
// ---------------------------------------------------------------------------
#if defined(__has_builtin)
#  if __has_builtin(__builtin_amdgcn_global_load_async_to_lds_b128)
#    define HAVE_ASYNC_LDS 1
#  endif
#endif

#ifdef HAVE_ASYNC_LDS
__device__ __forceinline__ void async_cp16(const void* g, void* l) {
    __builtin_amdgcn_global_load_async_to_lds_b128(
        (__attribute__((address_space(1))) v4i*)g,
        (__attribute__((address_space(3))) v4i*)l, 0, 0);
}
#  if __has_builtin(__builtin_amdgcn_s_wait_asynccnt)
#    define WAIT_ASYNCCNT(n) __builtin_amdgcn_s_wait_asynccnt(n)
#  else
#    define WAIT_ASYNCCNT(n) asm volatile("s_wait_asynccnt %0" :: "i"(n) : "memory")
#  endif
#endif

// ---------------------------------------------------------------------------
// fp32 -> f16 weight conversion
// ---------------------------------------------------------------------------
__global__ void k_f32_to_f16(const float* __restrict__ in, _Float16* __restrict__ out, int n) {
    int i = blockIdx.x * blockDim.x + threadIdx.x;
    if (i < n) out[i] = (_Float16)in[i];
}

// ---------------------------------------------------------------------------
// RMSNorm: one wave per token -> f16 (WMMA A operand)
// ---------------------------------------------------------------------------
__global__ void k_rmsnorm_f16(const float* __restrict__ x, const float* __restrict__ w,
                              _Float16* __restrict__ out) {
    int wave = threadIdx.x >> 5;
    int lane = threadIdx.x & 31;
    int tok  = blockIdx.x * (blockDim.x >> 5) + wave;
    if (tok >= NTOK) return;

    const float4* xr = (const float4*)(x + (size_t)tok * DMODEL);
    const float4* wr = (const float4*)w;
    float4 xv[4];
    float ss = 0.0f;
#pragma unroll
    for (int it = 0; it < 4; ++it) {
        xv[it] = xr[lane + it * 32];
        ss += xv[it].x * xv[it].x + xv[it].y * xv[it].y +
              xv[it].z * xv[it].z + xv[it].w * xv[it].w;
    }
#pragma unroll
    for (int m = 16; m >= 1; m >>= 1) ss += __shfl_xor(ss, m, 32);
    float scale = rsqrtf(ss * (1.0f / DMODEL) + 1e-5f);

    v4h* orow = (v4h*)(out + (size_t)tok * DMODEL);
#pragma unroll
    for (int it = 0; it < 4; ++it) {
        float4 wv = wr[lane + it * 32];
        v4h o;
        o[0] = (_Float16)(xv[it].x * scale * wv.x);
        o[1] = (_Float16)(xv[it].y * scale * wv.y);
        o[2] = (_Float16)(xv[it].z * scale * wv.z);
        o[3] = (_Float16)(xv[it].w * scale * wv.w);
        orow[lane + it * 32] = o;
    }
}

// ---------------------------------------------------------------------------
// A-fragment loader: two 16x32 fragments (one K-slab of 64) from global
// halves 0..7 -> K+akoff, halves 8..15 -> K+8+akoff (akoff = 16*(lane/16))
// ---------------------------------------------------------------------------
__device__ __forceinline__ void load_aslab(const _Float16* ap, int k0, F16Frag f[2]) {
#pragma unroll
    for (int kk = 0; kk < 2; ++kk) {
        const _Float16* p = ap + k0 + kk * 32;
        f[kk].u[0] = *(const v4u*)(p);
        f[kk].u[1] = *(const v4u*)(p + 8);
    }
}

#define ROWB  144                 // padded LDS row stride (128 data + 16 pad)
#define BUFB  (64 * ROWB)         // 9216 bytes per buffer

// One K-slab of WMMA work: batch 4 B-frag LDS loads per k-step, then 4 WMMAs
__device__ __forceinline__ void wmma_slab(const char* buf, const F16Frag a[2],
                                          v8f acc[4], int bl15, int bkob) {
#pragma unroll
    for (int kk = 0; kk < 2; ++kk) {
        const char* r0 = buf + (size_t)bl15 * ROWB + (kk << 6) + bkob;
        const char* r1 = r0 + 16 * ROWB;
        const char* r2 = r1 + 16 * ROWB;
        const char* r3 = r2 + 16 * ROWB;
        F16Frag b0, b1, b2, b3;
        b0.u[0] = *(const v4u*)(r0); b0.u[1] = *(const v4u*)(r0 + 16);
        b1.u[0] = *(const v4u*)(r1); b1.u[1] = *(const v4u*)(r1 + 16);
        b2.u[0] = *(const v4u*)(r2); b2.u[1] = *(const v4u*)(r2 + 16);
        b3.u[0] = *(const v4u*)(r3); b3.u[1] = *(const v4u*)(r3 + 16);
        acc[0] = __builtin_amdgcn_wmma_f32_16x16x32_f16(false, a[kk].v, false, b0.v, (short)0, acc[0], false, false);
        acc[1] = __builtin_amdgcn_wmma_f32_16x16x32_f16(false, a[kk].v, false, b1.v, (short)0, acc[1], false, false);
        acc[2] = __builtin_amdgcn_wmma_f32_16x16x32_f16(false, a[kk].v, false, b2.v, (short)0, acc[2], false, false);
        acc[3] = __builtin_amdgcn_wmma_f32_16x16x32_f16(false, a[kk].v, false, b3.v, (short)0, acc[3], false, false);
    }
}

// ---------------------------------------------------------------------------
// LDS-staged WMMA GEMM: C[M,N] = A[M,K]f16 * W[N,K]f16^T (+bias)(+softplus)(+add)
// Block: 256 threads = 8 waves; macro-tile 128(M) x 64(N); K-slab 64.
// Pairwise-unrolled slab loop: even slab -> buf0, odd slab -> buf1, with
// named A-fragment sets (no dynamic indexing -> no scratch spills).
// Requires nslab = K/64 even (true for K=512 and K=1024).
// ---------------------------------------------------------------------------
__global__ __launch_bounds__(256)
void k_gemm_wmma_lds(const _Float16* __restrict__ A, const _Float16* __restrict__ W,
                     const float* __restrict__ bias, const float* __restrict__ addsrc,
                     float* __restrict__ C, _Float16* __restrict__ dt16out,
                     int M, int N, int K, int act) {
    __shared__ __align__(16) char smem[2 * BUFB];

    const int tid  = threadIdx.x;
    const int lane = tid & 31;
    const int wave = tid >> 5;

    const int n0 = blockIdx.x << 6;                    // N tile (64 wide, block-shared)
    const int m0 = (blockIdx.y << 7) + (wave << 4);    // this wave's 16 M rows

    // staging coords: 64 rows x 128 bytes = 256 threads x 32 bytes
    const int srow  = tid >> 2;
    const int scolb = (tid & 3) << 5;
    const char* gsrc0 = (const char*)W + ((size_t)(n0 + srow) * K) * 2 + scolb;
    char* l0 = smem + srow * ROWB + scolb;             // dest in buf0
    char* l1 = l0 + BUFB;                              // dest in buf1

    // A fragment addressing
    const int akoff = (lane >> 4) << 4;                // 0 or 16
    const _Float16* ap = A + (size_t)(m0 + (lane & 15)) * K + akoff;

    // B fragment addressing within an LDS slab row
    const int bl15 = lane & 15;
    const int bkob = (lane >> 4) << 5;                 // byte offset 0 or 32

    v8f acc[4];
#pragma unroll
    for (int j = 0; j < 4; ++j)
#pragma unroll
        for (int i = 0; i < 8; ++i) acc[j][i] = 0.0f;

    const int nslab = K >> 6;                          // even: 8 or 16
    F16Frag aA[2], aB[2];
    load_aslab(ap, 0, aA);

#ifdef HAVE_ASYNC_LDS
    async_cp16(gsrc0,      l0);
    async_cp16(gsrc0 + 16, l0 + 16);

    for (int i = 0; i < nslab; i += 2) {
        // ---- even slab i (buf0); issue slab i+1 -> buf1 ahead ----
        {
            const char* g = gsrc0 + (size_t)(i + 1) * 128;
            async_cp16(g,      l1);
            async_cp16(g + 16, l1 + 16);
            load_aslab(ap, (i + 1) << 6, aB);
        }
        WAIT_ASYNCCNT(2);                              // slab i done, i+1 in flight
        __syncthreads();
        wmma_slab(smem, aA, acc, bl15, bkob);
        __syncthreads();

        // ---- odd slab i+1 (buf1); issue slab i+2 -> buf0 ahead ----
        if (i + 2 < nslab) {
            const char* g = gsrc0 + (size_t)(i + 2) * 128;
            async_cp16(g,      l0);
            async_cp16(g + 16, l0 + 16);
            load_aslab(ap, (i + 2) << 6, aA);
            WAIT_ASYNCCNT(2);
        } else {
            WAIT_ASYNCCNT(0);
        }
        __syncthreads();
        wmma_slab(smem + BUFB, aB, acc, bl15, bkob);
        __syncthreads();
    }
#else
    // fallback: regs + ds_store staging, same pairwise pipeline
    v4u t0 = *(const v4u*)(gsrc0), t1 = *(const v4u*)(gsrc0 + 16);
    *(v4u*)l0 = t0; *(v4u*)(l0 + 16) = t1;
    __syncthreads();

    for (int i = 0; i < nslab; i += 2) {
        {
            const char* g = gsrc0 + (size_t)(i + 1) * 128;
            t0 = *(const v4u*)(g); t1 = *(const v4u*)(g + 16);
            load_aslab(ap, (i + 1) << 6, aB);
        }
        wmma_slab(smem, aA, acc, bl15, bkob);
        __syncthreads();
        *(v4u*)l1 = t0; *(v4u*)(l1 + 16) = t1;
        __syncthreads();

        if (i + 2 < nslab) {
            const char* g = gsrc0 + (size_t)(i + 2) * 128;
            t0 = *(const v4u*)(g); t1 = *(const v4u*)(g + 16);
            load_aslab(ap, (i + 2) << 6, aA);
        }
        wmma_slab(smem + BUFB, aB, acc, bl15, bkob);
        __syncthreads();
        if (i + 2 < nslab) {
            *(v4u*)l0 = t0; *(v4u*)(l0 + 16) = t1;
            __syncthreads();
        }
    }
#endif

    // D layout: VGPR v -> (M = m0 + (lane>=16?8:0) + v, N = n0 + j*16 + (lane&15))
    const int mrow = m0 + ((lane >> 4) << 3);
    const int ncol = n0 + bl15;
#pragma unroll
    for (int j = 0; j < 4; ++j) {
        int n   = ncol + (j << 4);
        float b = bias ? bias[n] : 0.0f;
#pragma unroll
        for (int v = 0; v < 8; ++v) {
            int   m   = mrow + v;
            float val = acc[j][v] + b;
            if (act == 1) val = softplus_fast(val);
            if (addsrc)   val += addsrc[(size_t)m * N + n];
            C[(size_t)m * N + n] = val;
            if (dt16out && n < DTRANK)
                dt16out[(size_t)m * DTRANK + n] = (_Float16)val;
        }
    }
}

// ---------------------------------------------------------------------------
// Direct (no-LDS) WMMA GEMM for tiny K (dt_proj: K=32, one WMMA step)
// ---------------------------------------------------------------------------
__global__ void k_gemm_wmma_direct(const _Float16* __restrict__ A, const _Float16* __restrict__ W,
                                   const float* __restrict__ bias, const float* __restrict__ addsrc,
                                   float* __restrict__ C, int M, int N, int K, int act) {
    int lane   = threadIdx.x & 31;
    int wid    = blockIdx.x * (blockDim.x >> 5) + (threadIdx.x >> 5);
    int tilesN = N >> 6;
    int total  = (M >> 4) * tilesN;
    if (wid >= total) return;               // wave-uniform: EXEC all-ones for WMMA
    int m0 = (wid / tilesN) << 4;
    int n0 = (wid % tilesN) << 6;

    int akoff = (lane >> 4) << 4;
    int bkoff = (lane >> 4) << 4;
    const _Float16* ap  = A + (size_t)(m0 + (lane & 15)) * K + akoff;
    const _Float16* bp0 = W + (size_t)(n0 + (lane & 15)) * K + bkoff;

    v8f acc[4];
#pragma unroll
    for (int j = 0; j < 4; ++j)
#pragma unroll
        for (int i = 0; i < 8; ++i) acc[j][i] = 0.0f;

    for (int k0 = 0; k0 < K; k0 += 32) {
        F16Frag a;
        a.u[0] = *(const v4u*)(ap + k0);
        a.u[1] = *(const v4u*)(ap + k0 + 8);
        F16Frag b0, b1, b2, b3;
        const _Float16* p0 = bp0 + k0;
        const _Float16* p1 = p0 + (size_t)16 * K;
        const _Float16* p2 = p1 + (size_t)16 * K;
        const _Float16* p3 = p2 + (size_t)16 * K;
        b0.u[0] = *(const v4u*)(p0); b0.u[1] = *(const v4u*)(p0 + 8);
        b1.u[0] = *(const v4u*)(p1); b1.u[1] = *(const v4u*)(p1 + 8);
        b2.u[0] = *(const v4u*)(p2); b2.u[1] = *(const v4u*)(p2 + 8);
        b3.u[0] = *(const v4u*)(p3); b3.u[1] = *(const v4u*)(p3 + 8);
        acc[0] = __builtin_amdgcn_wmma_f32_16x16x32_f16(false, a.v, false, b0.v, (short)0, acc[0], false, false);
        acc[1] = __builtin_amdgcn_wmma_f32_16x16x32_f16(false, a.v, false, b1.v, (short)0, acc[1], false, false);
        acc[2] = __builtin_amdgcn_wmma_f32_16x16x32_f16(false, a.v, false, b2.v, (short)0, acc[2], false, false);
        acc[3] = __builtin_amdgcn_wmma_f32_16x16x32_f16(false, a.v, false, b3.v, (short)0, acc[3], false, false);
    }

    int mrow = m0 + ((lane >> 4) << 3);
    int ncol = n0 + (lane & 15);
#pragma unroll
    for (int j = 0; j < 4; ++j) {
        int n   = ncol + (j << 4);
        float b = bias ? bias[n] : 0.0f;
#pragma unroll
        for (int v = 0; v < 8; ++v) {
            int   m   = mrow + v;
            float val = acc[j][v] + b;
            if (act == 1) val = softplus_fast(val);
            if (addsrc)   val += addsrc[(size_t)m * N + n];
            C[(size_t)m * N + n] = val;
        }
    }
}

// ---------------------------------------------------------------------------
// Causal depthwise conv (8 taps along L) + SiLU -> fp32 + f16 copies
// ---------------------------------------------------------------------------
__global__ void k_conv_silu(const float* __restrict__ xz, const float* __restrict__ cw,
                            const float* __restrict__ cb, float* __restrict__ xcf,
                            _Float16* __restrict__ xc16) {
    int idx = blockIdx.x * blockDim.x + threadIdx.x;
    if (idx >= SB * SL * DINNER) return;
    int d = idx & (DINNER - 1);
    int l = (idx >> 10) & (SL - 1);
    int b = idx >> 20;

    const float* wr = cw + d * DCONV;
    float s = cb[d];
    size_t base = ((size_t)b * SL) * (2 * DINNER) + d;
#pragma unroll
    for (int j = 0; j < DCONV; ++j) {
        int ll = l + j - (DCONV - 1);
        if (ll >= 0) s += xz[base + (size_t)ll * (2 * DINNER)] * wr[j];
    }
    float v = s * sigmoid_fast(s);
    xcf[idx]  = v;
    xc16[idx] = (_Float16)v;
}

// ---------------------------------------------------------------------------
// Selective scan: one lane per (b,d,s); 16-lane shfl_xor reduction per step
// ---------------------------------------------------------------------------
__global__ void k_scan(const float* __restrict__ delta, const float* __restrict__ xcf,
                       const float* __restrict__ dbc,   const float* __restrict__ A_log,
                       const float* __restrict__ Dp,    float* __restrict__ yplus) {
    int gid = blockIdx.x * blockDim.x + threadIdx.x;   // (b*DINNER + d)*16 + s
    int s = gid & 15;
    int d = (gid >> 4) & (DINNER - 1);
    int b = gid >> 14;

    float Aval = -__expf(A_log[d * DSTATE + s]);
    float Dval = Dp[d];
    float h = 0.0f;
    int tokbase = b * SL;

    for (int t = 0; t < SL; ++t) {
        int tok = tokbase + t;
        float dlt = delta[(size_t)tok * DINNER + d];
        float xt  = xcf[(size_t)tok * DINNER + d];
        const float* row = dbc + (size_t)tok * (DTRANK + 2 * DSTATE);
        float bt = row[DTRANK + s];
        float ct = row[DTRANK + DSTATE + s];

        h = __expf(dlt * Aval) * h + (dlt * xt) * bt;
        float yv = h * ct;
        yv += __shfl_xor(yv, 1, 32);
        yv += __shfl_xor(yv, 2, 32);
        yv += __shfl_xor(yv, 4, 32);
        yv += __shfl_xor(yv, 8, 32);
        if (s == 0) yplus[(size_t)tok * DINNER + d] = yv + Dval * xt;
    }
}

// ---------------------------------------------------------------------------
// Gate: ya = (y + D*xc) * silu(z) -> f16
// ---------------------------------------------------------------------------
__global__ void k_gate(const float* __restrict__ yplus, const float* __restrict__ xz,
                       _Float16* __restrict__ ya16) {
    int idx = blockIdx.x * blockDim.x + threadIdx.x;
    if (idx >= NTOK * DINNER) return;
    int tok = idx >> 10;
    int d   = idx & (DINNER - 1);
    float z = xz[(size_t)tok * (2 * DINNER) + DINNER + d];
    float v = yplus[idx] * z * sigmoid_fast(z);
    ya16[idx] = (_Float16)v;
}

// ---------------------------------------------------------------------------
// Host launch
// ---------------------------------------------------------------------------
static inline void launch_gemm_lds(const _Float16* A, const _Float16* W, const float* bias,
                                   const float* add, float* C, _Float16* dt16out,
                                   int M, int N, int K, int act, hipStream_t stream) {
    dim3 grid(N >> 6, M >> 7);              // N/64, M/128
    k_gemm_wmma_lds<<<grid, 256, 0, stream>>>(A, W, bias, add, C, dt16out, M, N, K, act);
}

extern "C" void kernel_launch(void* const* d_in, const int* in_sizes, int n_in,
                              void* d_out, int out_size, void* d_ws, size_t ws_size,
                              hipStream_t stream) {
    (void)in_sizes; (void)n_in; (void)out_size; (void)ws_size;

    const float* x_in       = (const float*)d_in[0];
    const float* norm_w     = (const float*)d_in[1];
    const float* in_proj_w  = (const float*)d_in[2];
    const float* conv_w     = (const float*)d_in[3];
    const float* conv_b     = (const float*)d_in[4];
    const float* x_proj_w   = (const float*)d_in[5];
    const float* dt_proj_w  = (const float*)d_in[6];
    const float* dt_proj_b  = (const float*)d_in[7];
    const float* A_log      = (const float*)d_in[8];
    const float* D_param    = (const float*)d_in[9];
    const float* out_proj_w = (const float*)d_in[10];

    char*  base = (char*)d_ws;
    size_t off  = 0;
    auto wsalloc = [&](size_t bytes) -> void* {
        void* r = base + off;
        off = (off + bytes + 255) & ~(size_t)255;
        return r;
    };

    const int NIN  = NLAYERS * 2 * DINNER * DMODEL;
    const int NXP  = NLAYERS * (DTRANK + 2 * DSTATE) * DINNER;
    const int NDT  = NLAYERS * DINNER * DTRANK;
    const int NOUT = NLAYERS * DMODEL * DINNER;

    _Float16* w16_in  = (_Float16*)wsalloc((size_t)NIN  * 2);
    _Float16* w16_x   = (_Float16*)wsalloc((size_t)NXP  * 2);
    _Float16* w16_dt  = (_Float16*)wsalloc((size_t)NDT  * 2);
    _Float16* w16_out = (_Float16*)wsalloc((size_t)NOUT * 2);

    _Float16* h16   = (_Float16*)wsalloc((size_t)NTOK * DMODEL * 2);
    float*    xz    = (float*)   wsalloc((size_t)NTOK * 2 * DINNER * 4);
    float*    xcf   = (float*)   wsalloc((size_t)NTOK * DINNER * 4);
    _Float16* xc16  = (_Float16*)wsalloc((size_t)NTOK * DINNER * 2);
    float*    dbc   = (float*)   wsalloc((size_t)NTOK * (DTRANK + 2 * DSTATE) * 4);
    _Float16* dt16  = (_Float16*)wsalloc((size_t)NTOK * DTRANK * 2);
    float*    dlt   = (float*)   wsalloc((size_t)NTOK * DINNER * 4);
    float*    yplus = (float*)   wsalloc((size_t)NTOK * DINNER * 4);
    _Float16* ya16  = (_Float16*)wsalloc((size_t)NTOK * DINNER * 2);
    float*    xb0   = (float*)   wsalloc((size_t)NTOK * DMODEL * 4);
    float*    xb1   = (float*)   wsalloc((size_t)NTOK * DMODEL * 4);

    k_f32_to_f16<<<(NIN  + 255) / 256, 256, 0, stream>>>(in_proj_w,  w16_in,  NIN);
    k_f32_to_f16<<<(NXP  + 255) / 256, 256, 0, stream>>>(x_proj_w,   w16_x,   NXP);
    k_f32_to_f16<<<(NDT  + 255) / 256, 256, 0, stream>>>(dt_proj_w,  w16_dt,  NDT);
    k_f32_to_f16<<<(NOUT + 255) / 256, 256, 0, stream>>>(out_proj_w, w16_out, NOUT);

    const int nElem = NTOK * DINNER;

    for (int l = 0; l < NLAYERS; ++l) {
        const float* xin  = (l == 0) ? x_in : ((l & 1) ? xb0 : xb1);
        float*       xout = (l == NLAYERS - 1) ? (float*)d_out : ((l & 1) ? xb1 : xb0);

        // 1) RMSNorm -> f16
        k_rmsnorm_f16<<<NTOK / 8, 256, 0, stream>>>(xin, norm_w + l * DMODEL, h16);

        // 2) in_proj: [4096,512] x [2048,512]^T -> xz  (LDS-staged WMMA)
        launch_gemm_lds(h16, w16_in + (size_t)l * 2 * DINNER * DMODEL,
                        nullptr, nullptr, xz, nullptr, NTOK, 2 * DINNER, DMODEL, 0, stream);

        // 3) causal depthwise conv + SiLU
        k_conv_silu<<<(nElem + 255) / 256, 256, 0, stream>>>(
            xz, conv_w + l * DINNER * DCONV, conv_b + l * DINNER, xcf, xc16);

        // 4) x_proj: [4096,1024] x [64,1024]^T -> (dt|B|C), dt also packed to f16
        launch_gemm_lds(xc16, w16_x + (size_t)l * (DTRANK + 2 * DSTATE) * DINNER,
                        nullptr, nullptr, dbc, dt16, NTOK, DTRANK + 2 * DSTATE, DINNER, 0, stream);

        // 5) dt_proj + bias + softplus -> delta  (K=32: direct WMMA)
        {
            int total  = (NTOK >> 4) * (DINNER >> 6);
            int blocks = (total + 7) / 8;
            k_gemm_wmma_direct<<<blocks, 256, 0, stream>>>(
                dt16, w16_dt + (size_t)l * DINNER * DTRANK,
                dt_proj_b + l * DINNER, nullptr, dlt, NTOK, DINNER, DTRANK, 1);
        }

        // 6) selective scan -> y + D*xc
        k_scan<<<(SB * DINNER * DSTATE) / 256, 256, 0, stream>>>(
            dlt, xcf, dbc, A_log + (size_t)l * DINNER * DSTATE, D_param + l * DINNER, yplus);

        // 7) gate with silu(z) -> f16
        k_gate<<<(nElem + 255) / 256, 256, 0, stream>>>(yplus, xz, ya16);

        // 8) out_proj + residual -> next x (or d_out)  (LDS-staged WMMA)
        launch_gemm_lds(ya16, w16_out + (size_t)l * DMODEL * DINNER,
                        nullptr, xin, xout, nullptr, NTOK, DMODEL, DINNER, 0, stream);
    }
}
